// MultiHeadAttention_28398323761963
// MI455X (gfx1250) — matrix operand order repaired and checked by reading, not compiled, not run on hardware
//
#include <hip/hip_runtime.h>

#define S_LEN 2048
#define DM    1024
#define NHEAD 16
#define KD    64
#define BATCH 2
#define ROWS  (BATCH * S_LEN)   // 4096

typedef __attribute__((ext_vector_type(16))) __bf16 v16bf;
typedef __attribute__((ext_vector_type(8)))  float  v8f;

typedef int v4i __attribute__((vector_size(16)));
typedef __attribute__((address_space(1))) v4i g_v4i;   // global (device) int4
typedef __attribute__((address_space(3))) v4i l_v4i;   // LDS int4

union Frag16 {
  v16bf v;
  unsigned int u[8];
};

// ---------------------------------------------------------------------------
// gfx1250 async global->LDS copy (ASYNCcnt path), with synchronous fallback
// so the file compiles on toolchains where the builtin is absent.
// Signature (confirmed by clang diagnostic): (v4i __device__*, v4i __shared__*,
// imm offset, imm cpol)
// ---------------------------------------------------------------------------
#if __has_builtin(__builtin_amdgcn_global_load_async_to_lds_b128)
#define HAVE_ASYNC 1
__device__ __forceinline__ void async_copy_b128(const void* g, void* l) {
  __builtin_amdgcn_global_load_async_to_lds_b128((g_v4i*)g, (l_v4i*)l, 0, 0);
}
#else
#define HAVE_ASYNC 0
__device__ __forceinline__ void async_copy_b128(const void* g, void* l) {
  *(uint4*)l = *(const uint4*)g;
}
#endif

__device__ __forceinline__ void wait_async_all() {
#if HAVE_ASYNC
#if __has_builtin(__builtin_amdgcn_s_wait_asynccnt)
  __builtin_amdgcn_s_wait_asynccnt(0);
#else
  asm volatile("s_wait_asynccnt 0x0" ::: "memory");
#endif
#endif
}

__device__ __forceinline__ unsigned short f2bf(float f) {
  union { float f; unsigned int u; } x;
  x.f = f;
  unsigned int u = x.u;
  u += 0x7fffu + ((u >> 16) & 1u);   // round-to-nearest-even
  return (unsigned short)(u >> 16);
}

// A-matrix fragment: 16x32 bf16 tile, LDS row-major [row][stride].
// Lane l<16 holds row l with K = {0..7, 16..23}; lane l+16: K = {8..15, 24..31}.
__device__ __forceinline__ v16bf load_fragA(const unsigned short* p, int stride) {
  const int lane = threadIdx.x & 31;
  const int l = lane & 15, hi = lane >> 4;
  const unsigned short* rp = p + l * stride + 8 * hi;
  Frag16 f;
#pragma unroll
  for (int q = 0; q < 4; ++q) {
    f.u[q]     = *(const unsigned int*)(rp + 2 * q);
    f.u[q + 4] = *(const unsigned int*)(rp + 2 * q + 16);
  }
  return f.v;
}

// B-matrix fragment: 32x16 bf16. LDS holds B^T row-major [N][stride].
// Lane l<16 holds column N=l with K = {0..7, 16..23}; lane l+16: K = {8..15, 24..31}.
__device__ __forceinline__ v16bf load_fragB(const unsigned short* p, int stride) {
  const int lane = threadIdx.x & 31;
  const int l = lane & 15, hi = lane >> 4;
  const unsigned short* rp = p + l * stride + 8 * hi;
  Frag16 f;
#pragma unroll
  for (int q = 0; q < 4; ++q) {
    f.u[q]     = *(const unsigned int*)(rp + 2 * q);
    f.u[q + 4] = *(const unsigned int*)(rp + 2 * q + 16);
  }
  return f.v;
}

__device__ __forceinline__ v8f wmma_bf16(v16bf a, v16bf b, v8f c) {
  return __builtin_amdgcn_wmma_f32_16x16x32_bf16(false, a, false, b, (short)0, c,
                                                 false, false);
}

// Stage a 128-row x 32-ushort bf16 tile (global row stride in ushorts) into LDS.
// 512 x 16B chunks, 2 per thread (256 threads).
__device__ __forceinline__ void stage_tile(const unsigned short* g,
                                           size_t g_stride,
                                           unsigned short* l) {
  const int tid = threadIdx.x;
#pragma unroll
  for (int t = 0; t < 2; ++t) {
    const int c = tid + t * 256;       // 0..511
    const int row = c >> 2;
    const int off = (c & 3) * 8;       // ushorts
    async_copy_b128(g + (size_t)row * g_stride + off, l + row * 32 + off);
  }
}

// ---------------------------------------------------------------------------
// Kernel 0: one-time bf16 conversion pass.
//   xb  [4096,1024]  = bf16(x)
//   waT [3072,1024]  = bf16(w_attn^T)   (so GEMM B-tiles are row-major [n][k])
//   wpT [1024,1024]  = bf16(w_proj^T)
// ---------------------------------------------------------------------------
#define N_X  (ROWS * DM)            // 4194304
#define N_WA (DM * 3 * DM)          // 3145728
#define N_WP (DM * DM)              // 1048576

__global__ __launch_bounds__(256)
void convert_bf16_kernel(const float* __restrict__ x,
                         const float* __restrict__ w_attn,
                         const float* __restrict__ w_proj,
                         unsigned short* __restrict__ xb,
                         unsigned short* __restrict__ waT,
                         unsigned short* __restrict__ wpT) {
  const size_t stride = (size_t)gridDim.x * 256;
  for (size_t i = (size_t)blockIdx.x * 256 + threadIdx.x;
       i < (size_t)(N_X + N_WA + N_WP); i += stride) {
    if (i < N_X) {
      xb[i] = f2bf(x[i]);
    } else if (i < (size_t)(N_X + N_WA)) {
      const size_t j = i - N_X;
      const int k = (int)(j / (3 * DM));
      const int n = (int)(j % (3 * DM));
      waT[(size_t)n * DM + k] = f2bf(w_attn[j]);
    } else {
      const size_t j = i - N_X - N_WA;
      const int k = (int)(j >> 10);
      const int n = (int)(j & (DM - 1));
      wpT[(size_t)n * DM + k] = f2bf(w_proj[j]);
    }
  }
}

// ---------------------------------------------------------------------------
// Kernel 1: qkv = x @ w_attn + b_attn, async double-buffered bf16 GEMM.
// Epilogue scatters (reference split order v,q,k):
//   chunk 0 -> presentV (f32) + vbT bf16 [B,H,64,S]
//   chunk 1 -> Q bf16 [B,H,S,64], pre-scaled by 1/sqrt(64)
//   chunk 2 -> presentK (f32) + kb  bf16 [B,H,S,64]
// ---------------------------------------------------------------------------
__global__ __launch_bounds__(256)
void qkv_gemm_kernel(const unsigned short* __restrict__ xb,
                     const unsigned short* __restrict__ waT,
                     const float* __restrict__ bias,
                     float* __restrict__ presentK, float* __restrict__ presentV,
                     unsigned short* __restrict__ qbf,
                     unsigned short* __restrict__ kb,
                     unsigned short* __restrict__ vbT) {
  __shared__ unsigned short As[2][128 * 32];   // [m][k]
  __shared__ unsigned short Bs[2][128 * 32];   // B^T: [n][k]

  const int tid  = threadIdx.x;
  const int lane = tid & 31;
  const int wave = tid >> 5;
  const int wm = wave & 3;
  const int wn = wave >> 2;
  const int m0 = blockIdx.y * 128;
  const int n0 = blockIdx.x * 128;
  const int ln = lane & 15, hi = lane >> 4;

  const unsigned short* ag = xb  + (size_t)m0 * DM;
  const unsigned short* bg = waT + (size_t)n0 * DM;

  const v8f vzero = {};
  v8f acc[2][4];
#pragma unroll
  for (int i = 0; i < 2; ++i)
#pragma unroll
    for (int j = 0; j < 4; ++j) acc[i][j] = vzero;

  stage_tile(ag, DM, As[0]);
  stage_tile(bg, DM, Bs[0]);

  const int NK = DM / 32;
  for (int kt = 0; kt < NK; ++kt) {
    const int cur = kt & 1;
    wait_async_all();
    __syncthreads();
    if (kt + 1 < NK) {
      stage_tile(ag + (kt + 1) * 32, DM, As[cur ^ 1]);
      stage_tile(bg + (kt + 1) * 32, DM, Bs[cur ^ 1]);
    }
    v16bf a[2], b[4];
#pragma unroll
    for (int i = 0; i < 2; ++i) a[i] = load_fragA(&As[cur][(wm * 32 + i * 16) * 32], 32);
#pragma unroll
    for (int j = 0; j < 4; ++j) b[j] = load_fragB(&Bs[cur][(wn * 64 + j * 16) * 32], 32);
#pragma unroll
    for (int i = 0; i < 2; ++i)
#pragma unroll
      for (int j = 0; j < 4; ++j) acc[i][j] = wmma_bf16(a[i], b[j], acc[i][j]);
  }

  // epilogue: bias + scatter into head-major tensors
#pragma unroll
  for (int j = 0; j < 4; ++j) {
    const int n_g = n0 + wn * 64 + j * 16 + ln;
    const float bj = bias[n_g];
    const int chunk = n_g >> 10;       // 0=V 1=Q 2=K
    const int c = n_g & (DM - 1);
    const int h = c >> 6;
    const int d = c & 63;
#pragma unroll
    for (int i = 0; i < 2; ++i) {
#pragma unroll
      for (int r = 0; r < 8; ++r) {
        const int m_g  = m0 + wm * 32 + i * 16 + r + 8 * hi;
        const int bIdx = m_g >> 11;
        const int sIdx = m_g & (S_LEN - 1);
        const int bh   = bIdx * NHEAD + h;
        const float val = acc[i][j][r] + bj;
        const size_t hidx = ((size_t)bh * S_LEN + sIdx) * KD + d;
        if (chunk == 0) {
          presentV[hidx] = val;
          vbT[((size_t)bh * KD + d) * S_LEN + sIdx] = f2bf(val);
        } else if (chunk == 1) {
          qbf[hidx] = f2bf(val * 0.125f);   // 1/sqrt(64)
        } else {
          presentK[hidx] = val;
          kb[hidx] = f2bf(val);
        }
      }
    }
  }
}

// ---------------------------------------------------------------------------
// Kernel 2: causal flash attention, async double-buffered K/V stream.
// One block = one (b,h) x 128 query rows; each of 8 waves owns 16 rows.
// ---------------------------------------------------------------------------
__device__ __forceinline__ void stage_kv(const unsigned short* kg,
                                         const unsigned short* vtg,
                                         unsigned short* ks, unsigned short* vs) {
  const int tid = threadIdx.x;
#pragma unroll
  for (int t = 0; t < 2; ++t) {        // Ks: 8KB linear ([key][d] contiguous)
    const int c = tid + t * 256;       // 0..511
    async_copy_b128(kg + c * 8, ks + c * 8);
  }
#pragma unroll
  for (int t = 0; t < 2; ++t) {        // Vs: [d][key], row stride S_LEN in global
    const int c = tid + t * 256;
    const int d = c >> 2, off = (c & 3) * 8;
    async_copy_b128(vtg + (size_t)d * S_LEN + off, vs + d * 64 + off);
  }
}

__global__ __launch_bounds__(256)
void flash_attn_kernel(const unsigned short* __restrict__ qbf,
                       const unsigned short* __restrict__ kb,
                       const unsigned short* __restrict__ vbT,
                       unsigned short* __restrict__ att) {
  __shared__ unsigned short Qs[128 * 64];        // [q][d] bf16, pre-scaled
  __shared__ unsigned short Ks[2][64 * 64];      // [key][d]  == B^T for Q@K^T
  __shared__ unsigned short Vs[2][64 * 64];      // [d][key]  == B^T for P@V
  __shared__ unsigned short Ps[8][16 * 64];      // per-wave P, [m][key]

  const int tid  = threadIdx.x;
  const int lane = tid & 31;
  const int wave = tid >> 5;
  const int ln = lane & 15, hi = lane >> 4;
  const int bh    = blockIdx.y;          // b*16 + h
  const int qbase = blockIdx.x * 128;

  // stage Q once: contiguous 16KB, 1024 chunks, 4 per thread
  {
    const unsigned short* qg = qbf + ((size_t)bh * S_LEN + qbase) * KD;
#pragma unroll
    for (int t = 0; t < 4; ++t) {
      const int c = tid + t * 256;
      async_copy_b128(qg + c * 8, Qs + c * 8);
    }
  }

  const unsigned short* kbase_p = kb  + (size_t)bh * S_LEN * KD;
  const unsigned short* vbase_p = vbT + (size_t)bh * KD * S_LEN;

  const v8f vzero = {};
  v8f oacc[4];
#pragma unroll
  for (int j = 0; j < 4; ++j) oacc[j] = vzero;
  float mrun[8], lrun[8];
#pragma unroll
  for (int r = 0; r < 8; ++r) { mrun[r] = -1e30f; lrun[r] = 0.0f; }

  const int nkt = qbase / 64 + 2;        // only tiles with key_base <= q_max
  stage_kv(kbase_p, vbase_p, Ks[0], Vs[0]);

  for (int kt = 0; kt < nkt; ++kt) {
    const int cur = kt & 1;
    const int kbcur = kt * 64;
    wait_async_all();
    __syncthreads();
    if (kt + 1 < nkt)
      stage_kv(kbase_p + (size_t)(kbcur + 64) * KD, vbase_p + (kbcur + 64),
               Ks[cur ^ 1], Vs[cur ^ 1]);

    // ---- scores: S = (Q*scale) @ K^T, K-dim = 64 ----
    v8f sacc[4];
#pragma unroll
    for (int j = 0; j < 4; ++j) sacc[j] = vzero;
#pragma unroll
    for (int kc = 0; kc < 64; kc += 32) {
      const v16bf aq = load_fragA(&Qs[(wave * 16) * 64 + kc], 64);
#pragma unroll
      for (int j = 0; j < 4; ++j) {
        const v16bf bk = load_fragB(&Ks[cur][(j * 16) * 64 + kc], 64);
        sacc[j] = wmma_bf16(aq, bk, sacc[j]);
      }
    }

    // ---- online softmax; C layout: row = r + 8*hi, col = ln + 16*j ----
#pragma unroll
    for (int r = 0; r < 8; ++r) {
      const int q_g = qbase + wave * 16 + r + 8 * hi;
      float sv[4];
      float mx = -1e30f;
#pragma unroll
      for (int j = 0; j < 4; ++j) {
        float sc = sacc[j][r];
        if (kbcur + j * 16 + ln > q_g) sc = -1e30f;   // causal mask
        sv[j] = sc;
        mx = fmaxf(mx, sc);
      }
#pragma unroll
      for (int off = 1; off < 16; off <<= 1)
        mx = fmaxf(mx, __shfl_xor(mx, off, 32));
      const float mnew  = fmaxf(mrun[r], mx);
      const float alpha = __expf(mrun[r] - mnew);
      mrun[r] = mnew;
      float rs = 0.0f;
#pragma unroll
      for (int j = 0; j < 4; ++j) {
        const float p = __expf(sv[j] - mnew);
        rs += p;
        Ps[wave][(r + 8 * hi) * 64 + j * 16 + ln] = f2bf(p);
      }
#pragma unroll
      for (int off = 1; off < 16; off <<= 1)
        rs += __shfl_xor(rs, off, 32);
      lrun[r] = lrun[r] * alpha + rs;
#pragma unroll
      for (int j = 0; j < 4; ++j) oacc[j][r] *= alpha;
    }

    // ---- O += P @ V, K-dim = 64 keys ----
#pragma unroll
    for (int kc = 0; kc < 64; kc += 32) {
      const v16bf ap = load_fragA(&Ps[wave][kc], 64);
#pragma unroll
      for (int j = 0; j < 4; ++j) {
        const v16bf bv = load_fragB(&Vs[cur][(j * 16) * 64 + kc], 64);
        oacc[j] = wmma_bf16(ap, bv, oacc[j]);
      }
    }
  }

  // normalize and store attention output as bf16 [B,S,DM]
  const int b = bh >> 4, h = bh & 15;
#pragma unroll
  for (int j = 0; j < 4; ++j) {
#pragma unroll
    for (int r = 0; r < 8; ++r) {
      const int srow = qbase + wave * 16 + r + 8 * hi;
      const float val = oacc[j][r] / lrun[r];
      att[(size_t)(b * S_LEN + srow) * DM + h * KD + j * 16 + ln] = f2bf(val);
    }
  }
}

// ---------------------------------------------------------------------------
// Kernel 3: out = att @ w_proj + b_proj, async double-buffered bf16 GEMM.
// ---------------------------------------------------------------------------
__global__ __launch_bounds__(256)
void proj_gemm_kernel(const unsigned short* __restrict__ att,
                      const unsigned short* __restrict__ wpT,
                      const float* __restrict__ bias,
                      float* __restrict__ out) {
  __shared__ unsigned short As[2][128 * 32];
  __shared__ unsigned short Bs[2][128 * 32];

  const int tid  = threadIdx.x;
  const int lane = tid & 31;
  const int wave = tid >> 5;
  const int wm = wave & 3;
  const int wn = wave >> 2;
  const int m0 = blockIdx.y * 128;
  const int n0 = blockIdx.x * 128;
  const int ln = lane & 15, hi = lane >> 4;

  const unsigned short* ag = att + (size_t)m0 * DM;
  const unsigned short* bg = wpT + (size_t)n0 * DM;

  const v8f vzero = {};
  v8f acc[2][4];
#pragma unroll
  for (int i = 0; i < 2; ++i)
#pragma unroll
    for (int j = 0; j < 4; ++j) acc[i][j] = vzero;

  stage_tile(ag, DM, As[0]);
  stage_tile(bg, DM, Bs[0]);

  const int NK = DM / 32;
  for (int kt = 0; kt < NK; ++kt) {
    const int cur = kt & 1;
    wait_async_all();
    __syncthreads();
    if (kt + 1 < NK) {
      stage_tile(ag + (kt + 1) * 32, DM, As[cur ^ 1]);
      stage_tile(bg + (kt + 1) * 32, DM, Bs[cur ^ 1]);
    }
    v16bf a[2], b[4];
#pragma unroll
    for (int i = 0; i < 2; ++i) a[i] = load_fragA(&As[cur][(wm * 32 + i * 16) * 32], 32);
#pragma unroll
    for (int j = 0; j < 4; ++j) b[j] = load_fragB(&Bs[cur][(wn * 64 + j * 16) * 32], 32);
#pragma unroll
    for (int i = 0; i < 2; ++i)
#pragma unroll
      for (int j = 0; j < 4; ++j) acc[i][j] = wmma_bf16(a[i], b[j], acc[i][j]);
  }

#pragma unroll
  for (int j = 0; j < 4; ++j) {
    const int n_g = n0 + wn * 64 + j * 16 + ln;
    const float bj = bias[n_g];
#pragma unroll
    for (int i = 0; i < 2; ++i) {
#pragma unroll
      for (int r = 0; r < 8; ++r) {
        const int m_g = m0 + wm * 32 + i * 16 + r + 8 * hi;
        out[(size_t)m_g * DM + n_g] = acc[i][j][r] + bj;
      }
    }
  }
}

extern "C" void kernel_launch(void* const* d_in, const int* in_sizes, int n_in,
                              void* d_out, int out_size, void* d_ws, size_t ws_size,
                              hipStream_t stream) {
  (void)in_sizes; (void)n_in; (void)out_size; (void)ws_size;
  const float* x      = (const float*)d_in[0];
  // d_in[1] = causal mask tensor: unused (mask applied analytically)
  const float* w_attn = (const float*)d_in[2];
  const float* b_attn = (const float*)d_in[3];
  const float* w_proj = (const float*)d_in[4];
  const float* b_proj = (const float*)d_in[5];

  float* out = (float*)d_out;                               // [B,S,DM]
  float* presentK = out + (size_t)ROWS * DM;                // present[0]: [B,H,S,KD]
  float* presentV = presentK + (size_t)BATCH * NHEAD * S_LEN * KD;  // present[1]

  unsigned short* qbf = (unsigned short*)d_ws;              // [B,H,S,KD] bf16
  unsigned short* att = qbf + (size_t)BATCH * NHEAD * S_LEN * KD;   // [B,S,DM]
  unsigned short* xb  = att + (size_t)ROWS * DM;            // [B*S,DM] bf16
  unsigned short* waT = xb  + (size_t)N_X;                  // [3DM,DM] bf16
  unsigned short* wpT = waT + (size_t)N_WA;                 // [DM,DM]  bf16
  unsigned short* kbb = wpT + (size_t)N_WP;                 // [B,H,S,KD] bf16
  unsigned short* vbT = kbb + (size_t)BATCH * NHEAD * S_LEN * KD;   // [B,H,KD,S]

  convert_bf16_kernel<<<8192, 256, 0, stream>>>(x, w_attn, w_proj, xb, waT, wpT);
  qkv_gemm_kernel<<<dim3((3 * DM) / 128, ROWS / 128), 256, 0, stream>>>(
      xb, waT, b_attn, presentK, presentV, qbf, kbb, vbT);
  flash_attn_kernel<<<dim3(S_LEN / 128, BATCH * NHEAD), 256, 0, stream>>>(
      qbf, kbb, vbT, att);
  proj_gemm_kernel<<<dim3(DM / 128, ROWS / 128), 256, 0, stream>>>(
      att, wpT, b_proj, out);
}
